// SynapForge100M_33363305955334
// MI455X (gfx1250) — compile-verified, matrix operand order checked by reading, not run on hardware
//
#include <hip/hip_runtime.h>
#include <hip/hip_bf16.h>
#include <math.h>
#include <stdint.h>

// ---- model constants ----
#define D_    512
#define T_    256
#define B_    4
#define H_    4096
#define L_    10
#define LOOP_ 4
#define V_    50257
#define BT_   (B_ * T_)      // 1024
#define THR_  0.05f
#define EPS_  1e-6f

typedef __attribute__((ext_vector_type(16))) __bf16 v16bf;
typedef __attribute__((ext_vector_type(8)))  __bf16 v8bf;
typedef __attribute__((ext_vector_type(8)))  float  v8f;

__device__ __forceinline__ v16bf cat16(v8bf lo, v8bf hi) {
  return __builtin_shufflevector(lo, hi, 0, 1, 2, 3, 4, 5, 6, 7,
                                 8, 9, 10, 11, 12, 13, 14, 15);
}

// ---------------------------------------------------------------------------
// bf16 fragment loaders (ISA 7.12.2 16-bit 16x32 A layout; B for C = A @ W^T
// reads row n of W(N,K) with the identical pattern). Two b128 loads per frag.
// ---------------------------------------------------------------------------
__device__ __forceinline__ v16bf frag_ld(const __bf16* __restrict__ base, int ld,
                                         int r, int k0, int lane) {
  const int kb = k0 + ((lane >> 4) << 3);
  const __bf16* p = base + (size_t)r * ld + kb;
  return cat16(*(const v8bf*)p, *(const v8bf*)(p + 16));
}

__device__ __forceinline__ v16bf frag_ld_guard(const __bf16* __restrict__ base, int ld,
                                               int r, int rmax, int k0, int lane) {
  if (r < rmax) return frag_ld(base, ld, r, k0, lane);
  v16bf z;
#pragma unroll
  for (int i = 0; i < 16; ++i) z[i] = (__bf16)0.f;
  return z;
}

__device__ __forceinline__ v8f wmma_bf(v16bf a, v16bf b, v8f c) {
  return __builtin_amdgcn_wmma_f32_16x16x32_bf16(false, a, false, b,
                                                 (short)0, c, false, false);
}

__device__ __forceinline__ float sigmoidf_(float x) { return 1.f / (1.f + __expf(-x)); }

// ---------------------------------------------------------------------------
// Async global->LDS staging of a 32-row x 32-k bf16 A tile (2 KB).
// Threads 0..127 each issue one 16-byte GLOBAL_LOAD_ASYNC_TO_LDS_B128
// (tracked by ASYNCcnt); consumers wait with s_wait_asynccnt + barrier.
// LDS layout: row-major, 64 B per row. NOTE: assumes the (single) __shared__
// array of the caller sits at LDS offset 0.
// ---------------------------------------------------------------------------
__device__ __forceinline__ void stage_a_async(const __bf16* __restrict__ A, int K,
                                              int rowbase, int k0,
                                              uint32_t lds_byte_base, int tid) {
  if (tid < 128) {
    const int r = tid >> 2, c = tid & 3;
    const uint32_t loff = lds_byte_base + (uint32_t)(r * 64 + c * 16);
    const __bf16* g = A + (size_t)(rowbase + r) * K + k0 + c * 8;
    asm volatile("global_load_async_to_lds_b128 %0, %1, off"
                 :: "v"(loff), "v"((unsigned long long)(uintptr_t)g)
                 : "memory");
  }
}
#define WAIT_ASYNC() asm volatile("s_wait_asynccnt 0x0" ::: "memory")

// ---------------------------------------------------------------------------
// WMMA GEMM v2: C(MxN) = A(MxK)bf16 @ W(NxK)bf16^T  [+bias(N)] [+resid(MxN)]
// Block = 256 thr = 8 waves = 2 (M) x 4 (N); wave tile 16x64 (4 accumulators,
// A-fragment reused 4x). A tile double-buffered in LDS via async copies.
// M % 32 == 0, K % 32 == 0; N guarded (LM head edge).
// MODE: 0 plain, 1 +bias, 2 resid+acc.   C is fp32.
// ---------------------------------------------------------------------------
template <int MODE>
__global__ __launch_bounds__(256) void gemm16v2(const __bf16* __restrict__ A,
                                                const __bf16* __restrict__ W,
                                                const float* __restrict__ bias,
                                                const float* __restrict__ resid,
                                                float* __restrict__ C,
                                                int M, int N, int K) {
  __shared__ __bf16 atile[2][32 * 32];     // 2 x 2 KB, assumed at LDS offset 0
  const int tid = threadIdx.x, lane = tid & 31, wave = tid >> 5;
  const int wm = wave >> 2, wn = wave & 3;
  const int nblk = (N + 255) >> 8;
  const int bm = blockIdx.x / nblk, bn = blockIdx.x % nblk;
  const int rowbase = bm * 32;
  const int m0 = rowbase + wm * 16;
  const int n0 = bn * 256 + wn * 64;
  const int l15 = lane & 15, hi8 = (lane >> 4) << 3;

  stage_a_async(A, K, rowbase, 0, 0u, tid);
  WAIT_ASYNC();
  __syncthreads();

  v8f acc[4] = {};
  const int nk = K >> 5;
#pragma unroll 1
  for (int kt = 0; kt < nk; ++kt) {
    if (kt + 1 < nk)
      stage_a_async(A, K, rowbase, (kt + 1) << 5, (uint32_t)(((kt + 1) & 1) * 2048), tid);
    const __bf16* ap = &atile[kt & 1][(wm * 16 + l15) * 32 + hi8];
    const v16bf a = cat16(*(const v8bf*)ap, *(const v8bf*)(ap + 16));
#pragma unroll
    for (int s = 0; s < 4; ++s) {
      v16bf b = frag_ld_guard(W, K, n0 + s * 16 + l15, N, kt << 5, lane);
      acc[s] = wmma_bf(a, b, acc[s]);
    }
    WAIT_ASYNC();
    __syncthreads();
  }

#pragma unroll
  for (int s = 0; s < 4; ++s) {
    const int col = n0 + s * 16 + l15;
    if (col < N) {
#pragma unroll
      for (int j = 0; j < 8; ++j) {
        const int row = m0 + j + hi8;
        float v = acc[s][j];
        if (MODE == 1) v += bias[col];
        if (MODE == 2) v += resid[(size_t)row * N + col];
        C[(size_t)row * N + col] = v;
      }
    }
  }
}

// ---------------------------------------------------------------------------
// Dual-B fused GEMM v2: acc1 = A@W1^T, acc2 = A@W2^T sharing the A fragment.
// OP 0 (spiking gate): Cf32 = resid + acc1 * sigmoid(acc2 + bias2)
// OP 1 (FFN swiglu)  : Cbf16 = silu(acc1) * acc2
// N % 256 == 0 here (512 / 4096), so no column guards.
// ---------------------------------------------------------------------------
template <int OP>
__global__ __launch_bounds__(256) void gemm16v2_dual(const __bf16* __restrict__ A,
                                                     const __bf16* __restrict__ W1,
                                                     const __bf16* __restrict__ W2,
                                                     const float* __restrict__ bias2,
                                                     const float* __restrict__ resid,
                                                     void* __restrict__ Cout,
                                                     int M, int N, int K) {
  __shared__ __bf16 atile[2][32 * 32];
  const int tid = threadIdx.x, lane = tid & 31, wave = tid >> 5;
  const int wm = wave >> 2, wn = wave & 3;
  const int nblk = N >> 8;
  const int bm = blockIdx.x / nblk, bn = blockIdx.x % nblk;
  const int rowbase = bm * 32;
  const int m0 = rowbase + wm * 16;
  const int n0 = bn * 256 + wn * 64;
  const int l15 = lane & 15, hi8 = (lane >> 4) << 3;

  stage_a_async(A, K, rowbase, 0, 0u, tid);
  WAIT_ASYNC();
  __syncthreads();

  v8f acc1[4] = {}, acc2[4] = {};
  const int nk = K >> 5;
#pragma unroll 1
  for (int kt = 0; kt < nk; ++kt) {
    if (kt + 1 < nk)
      stage_a_async(A, K, rowbase, (kt + 1) << 5, (uint32_t)(((kt + 1) & 1) * 2048), tid);
    const __bf16* ap = &atile[kt & 1][(wm * 16 + l15) * 32 + hi8];
    const v16bf a = cat16(*(const v8bf*)ap, *(const v8bf*)(ap + 16));
#pragma unroll
    for (int s = 0; s < 4; ++s) {
      const int nr = n0 + s * 16 + l15;
      v16bf b1 = frag_ld(W1, K, nr, kt << 5, lane);
      v16bf b2 = frag_ld(W2, K, nr, kt << 5, lane);
      acc1[s] = wmma_bf(a, b1, acc1[s]);
      acc2[s] = wmma_bf(a, b2, acc2[s]);
    }
    WAIT_ASYNC();
    __syncthreads();
  }

#pragma unroll
  for (int s = 0; s < 4; ++s) {
    const int col = n0 + s * 16 + l15;
#pragma unroll
    for (int j = 0; j < 8; ++j) {
      const int row = m0 + j + hi8;
      if (OP == 0) {
        float o = resid[(size_t)row * N + col] +
                  acc1[s][j] * sigmoidf_(acc2[s][j] + bias2[col]);
        ((float*)Cout)[(size_t)row * N + col] = o;
      } else {
        float g = acc1[s][j];
        ((__bf16*)Cout)[(size_t)row * N + col] =
            (__bf16)(g * sigmoidf_(g) * acc2[s][j]);
      }
    }
  }
}

// ---------------------------------------------------------------------------
// Liquid/PLIF recurrent scan: one persistent 1024-thread workgroup (32 waves).
// h (16x512, rows 0..3 live) in LDS as bf16 (WMMA A) + fp32 state; per step
// pre = ain[t] + h @ wrec^T, tanh leaky update, PLIF charge/spike/reset.
// k-loop forced to unroll 1: with full unroll the compiler hoists all 16
// loop-invariant wrec B-fragments (128 VGPRs/lane) out of the t-loop and
// spills to scratch at 1024 thr/WG; re-reading L2-resident wrec is cheaper.
// ---------------------------------------------------------------------------
__global__ __launch_bounds__(1024) void liquid_scan_k(const float* __restrict__ ain,    // (BT,D) f32
                                                      const __bf16* __restrict__ wrecb, // (D,D) bf16
                                                      const float* __restrict__ ltau,   // (D)
                                                      const float* __restrict__ pw,     // scalar
                                                      __bf16* __restrict__ s_out) {     // (BT,D) bf16
  __shared__ __bf16 hbf[16 * D_];
  __shared__ float  hF[B_ * D_];
  __shared__ float  vF[B_ * D_];
  const int tid = threadIdx.x, lane = tid & 31, wave = tid >> 5;

  for (int i = tid; i < 16 * D_; i += 1024) hbf[i] = (__bf16)0.f;
  for (int i = tid; i < B_ * D_; i += 1024) { hF[i] = 0.f; vF[i] = 0.f; }

  const float decay = sigmoidf_(pw[0]);
  const int col = (wave << 4) + (lane & 15);
  const float invtau = 1.f / (log1pf(__expf(ltau[col])) + 1.f);
  const int hi = lane >> 4;
  __syncthreads();

#pragma unroll 1
  for (int t = 0; t < T_; ++t) {
    v8f acc = {};
#pragma unroll 1
    for (int k0 = 0; k0 < D_; k0 += 32) {
      const __bf16* ap = &hbf[(lane & 15) * D_ + k0 + (hi << 3)];
      const v16bf a = cat16(*(const v8bf*)ap, *(const v8bf*)(ap + 16));
      const v16bf b = frag_ld(wrecb, D_, col, k0, lane);
      acc = wmma_bf(a, b, acc);
    }
    __syncthreads();                       // reads of hbf complete
#pragma unroll
    for (int j = 0; j < 8; ++j) {
      const int row = j + (hi << 3);
      if (row < B_) {
        const size_t g = ((size_t)row * T_ + t) * D_ + col;
        float pre = acc[j] + ain[g];
        if (t + 1 < T_) __builtin_prefetch(&ain[g + D_], 0, 3);
        float h = hF[row * D_ + col];
        h = h + (tanhf(pre) - h) * invtau;
        float v = vF[row * D_ + col];
        v += decay * (h - v);
        float s = (v - THR_) > 0.f ? 1.f : 0.f;
        v -= s * THR_;
        hF[row * D_ + col] = h;
        vF[row * D_ + col] = v;
        hbf[row * D_ + col] = (__bf16)h;
        s_out[g] = (__bf16)s;
      }
    }
    __syncthreads();                       // h visible before next WMMA
  }
}

// ---------------------------------------------------------------------------
// Elementwise kernels
// ---------------------------------------------------------------------------
__global__ void embed_k(const int* __restrict__ ids, const float* __restrict__ tok,
                        const float* __restrict__ pos, float* __restrict__ x) {
  const int bt = blockIdx.x;
  const int t = bt & (T_ - 1);
  const int id = ids[bt];
  for (int d = threadIdx.x; d < D_; d += blockDim.x)
    x[(size_t)bt * D_ + d] = tok[(size_t)id * D_ + d] + pos[(size_t)t * D_ + d];
}

// RMSNorm producing a bf16 matrix (GEMM A operand)
__global__ void rmsnorm_bf_k(const float* __restrict__ x, const float* __restrict__ w,
                             __bf16* __restrict__ out) {
  const int row = blockIdx.x;
  const float* xr = x + (size_t)row * D_;
  __shared__ float red[256];
  float s = 0.f;
  for (int i = threadIdx.x; i < D_; i += 256) { float v = xr[i]; s += v * v; }
  red[threadIdx.x] = s;
  __syncthreads();
  for (int o = 128; o > 0; o >>= 1) {
    if (threadIdx.x < o) red[threadIdx.x] += red[threadIdx.x + o];
    __syncthreads();
  }
  const float inv = rsqrtf(red[0] * (1.f / D_) + EPS_);
  for (int i = threadIdx.x; i < D_; i += 256)
    out[(size_t)row * D_ + i] = (__bf16)(xr[i] * inv * w[i]);
}

__global__ void cvt_bf16_k(const float* __restrict__ in, __bf16* __restrict__ out, long n) {
  long i = (long)blockIdx.x * blockDim.x + threadIdx.x;
  const long stride = (long)gridDim.x * blockDim.x;
  for (; i < n; i += stride) out[i] = (__bf16)in[i];
}

__global__ void weff_bf_k(const float* __restrict__ syn, const unsigned char* __restrict__ mask,
                          __bf16* __restrict__ out, int n) {
  const int i = blockIdx.x * blockDim.x + threadIdx.x;
  if (i < n) out[i] = mask[i] ? (__bf16)syn[i] : (__bf16)0.f;
}

// ---------------------------------------------------------------------------
// Host orchestration
// ---------------------------------------------------------------------------
static inline int v2_blocks(int M, int N) { return (M / 32) * ((N + 255) >> 8); }
static inline int cvt_blocks(long n) {
  long b = (n + 255) / 256;
  return (int)(b > 16384 ? 16384 : b);
}

extern "C" void kernel_launch(void* const* d_in, const int* in_sizes, int n_in,
                              void* d_out, int out_size, void* d_ws, size_t ws_size,
                              hipStream_t stream) {
  (void)in_sizes; (void)n_in; (void)out_size; (void)ws_size;
  const int*   ids  = (const int*)d_in[0];
  const float* tok  = (const float*)d_in[1];
  const float* pos  = (const float*)d_in[2];
  const float* ln1  = (const float*)d_in[3];
  const float* win  = (const float*)d_in[4];
  const float* wrec = (const float*)d_in[5];
  const float* lb   = (const float*)d_in[6];
  const float* ltau = (const float*)d_in[7];
  const float* pw   = (const float*)d_in[8];
  const float* syn  = (const float*)d_in[9];
  const unsigned char* mask = (const unsigned char*)d_in[10];
  const float* gw   = (const float*)d_in[11];
  const float* gb   = (const float*)d_in[12];
  const float* ln2  = (const float*)d_in[13];
  const float* fg   = (const float*)d_in[14];
  const float* fu   = (const float*)d_in[15];
  const float* fd   = (const float*)d_in[16];
  const float* lnf  = (const float*)d_in[17];
  float* out = (float*)d_out;

  const int DD = D_ * D_, HD = H_ * D_;

  // ---- workspace carve-up (256 B aligned) ----
  uint8_t* w8 = (uint8_t*)d_ws;
  auto alloc = [&](size_t bytes) {
    uint8_t* p = w8;
    w8 += (bytes + 255) & ~(size_t)255;
    return p;
  };
  float*  x     = (float*)alloc((size_t)BT_ * D_ * 4);
  float*  x2    = (float*)alloc((size_t)BT_ * D_ * 4);
  float*  ainb  = (float*)alloc((size_t)BT_ * D_ * 4);
  __bf16* abuf  = (__bf16*)alloc((size_t)BT_ * D_ * 2);
  __bf16* sbf   = (__bf16*)alloc((size_t)BT_ * D_ * 2);
  __bf16* hbuf  = (__bf16*)alloc((size_t)BT_ * H_ * 2);
  __bf16* tokb  = (__bf16*)alloc((size_t)V_ * D_ * 2);
  __bf16* winb  = (__bf16*)alloc((size_t)DD * 2);
  __bf16* wrecb = (__bf16*)alloc((size_t)DD * 2);
  __bf16* weffb = (__bf16*)alloc((size_t)DD * 2);
  __bf16* gwb   = (__bf16*)alloc((size_t)DD * 2);
  __bf16* fgb   = (__bf16*)alloc((size_t)HD * 2);
  __bf16* fub   = (__bf16*)alloc((size_t)HD * 2);
  __bf16* fdb   = (__bf16*)alloc((size_t)HD * 2);

  // one-time: tied-embedding bf16 copy + input embedding
  cvt_bf16_k<<<cvt_blocks((long)V_ * D_), 256, 0, stream>>>(tok, tokb, (long)V_ * D_);
  embed_k<<<BT_, 256, 0, stream>>>(ids, tok, pos, x);

  for (int l = 0; l < L_; ++l) {
    const float* lb_l   = lb   + (size_t)l * D_;
    const float* ltau_l = ltau + (size_t)l * D_;
    const float* pw_l   = pw   + l;
    const float* gb_l   = gb   + (size_t)l * D_;
    const float* ln1_l  = ln1  + (size_t)l * D_;
    const float* ln2_l  = ln2  + (size_t)l * D_;

    // per-layer bf16 weight conversions (reused across LOOP_ iterations)
    cvt_bf16_k<<<cvt_blocks(DD), 256, 0, stream>>>(win + (size_t)l * DD, winb, DD);
    cvt_bf16_k<<<cvt_blocks(DD), 256, 0, stream>>>(wrec + (size_t)l * DD, wrecb, DD);
    cvt_bf16_k<<<cvt_blocks(DD), 256, 0, stream>>>(gw + (size_t)l * DD, gwb, DD);
    weff_bf_k<<<(DD + 255) / 256, 256, 0, stream>>>(syn + (size_t)l * DD,
                                                    mask + (size_t)l * DD, weffb, DD);
    cvt_bf16_k<<<cvt_blocks(HD), 256, 0, stream>>>(fg + (size_t)l * HD, fgb, HD);
    cvt_bf16_k<<<cvt_blocks(HD), 256, 0, stream>>>(fu + (size_t)l * HD, fub, HD);
    cvt_bf16_k<<<cvt_blocks(HD), 256, 0, stream>>>(fd + (size_t)l * HD, fdb, HD);

    for (int r = 0; r < LOOP_; ++r) {
      rmsnorm_bf_k<<<BT_, 256, 0, stream>>>(x, ln1_l, abuf);
      gemm16v2<1><<<v2_blocks(BT_, D_), 256, 0, stream>>>(abuf, winb, lb_l, nullptr,
                                                          ainb, BT_, D_, D_);
      liquid_scan_k<<<1, 1024, 0, stream>>>(ainb, wrecb, ltau_l, pw_l, sbf);
      gemm16v2_dual<0><<<v2_blocks(BT_, D_), 256, 0, stream>>>(sbf, weffb, gwb, gb_l,
                                                               x, (void*)x2, BT_, D_, D_);
      rmsnorm_bf_k<<<BT_, 256, 0, stream>>>(x2, ln2_l, abuf);
      gemm16v2_dual<1><<<v2_blocks(BT_, H_), 256, 0, stream>>>(abuf, fgb, fub, nullptr,
                                                               nullptr, (void*)hbuf,
                                                               BT_, H_, D_);
      gemm16v2<2><<<v2_blocks(BT_, D_), 256, 0, stream>>>(hbuf, fdb, nullptr, x2,
                                                          x, BT_, D_, H_);
    }
  }

  rmsnorm_bf_k<<<BT_, 256, 0, stream>>>(x, lnf, abuf);
  gemm16v2<0><<<v2_blocks(BT_, V_), 256, 0, stream>>>(abuf, tokb, nullptr, nullptr,
                                                      out, BT_, V_, D_);
}
